// IDMForwardSim_77360950935639
// MI455X (gfx1250) — matrix operand order repaired and checked by reading, not compiled, not run on hardware
//
#include <hip/hip_runtime.h>

typedef __attribute__((ext_vector_type(16))) _Float16 v16h;
typedef __attribute__((ext_vector_type(8)))  _Float16 v8h;
typedef __attribute__((ext_vector_type(8)))  float    v8f;
typedef __attribute__((ext_vector_type(4)))  float    v4f;
typedef __attribute__((ext_vector_type(4)))  unsigned int v4u;

#define T_STEPS  40
#define D_IN     128
#define H_DIM    100
#define N_PAD    48      // 40 -> 3 tiles of 16
#define K_PAD    136     // 128 + 8 halves pad: LDS row stride 272B kills bank conflicts
#define ATT_TEMP 5.0f
#define ACT_CLIP 3.5f

// ---------------------------------------------------------------------------
// Kernel 0: fuse the two dense layers (no nonlinearity between them):
//   Wc  = W1 @ W2          (128 x 40, stored N-major f16, padded [48][136])
//   bc  = b1 @ W2 + b2     (f32, padded to 48)
// ---------------------------------------------------------------------------
__global__ void idm_prep_weights(const float* __restrict__ W1, const float* __restrict__ b1,
                                 const float* __restrict__ W2, const float* __restrict__ b2,
                                 _Float16* __restrict__ WcT, float* __restrict__ bc) {
  const int idx = blockIdx.x * blockDim.x + threadIdx.x;
  if (idx >= N_PAD * K_PAD) return;
  const int n = idx / K_PAD;
  const int k = idx % K_PAD;
  float acc = 0.0f;
  if (n < T_STEPS && k < D_IN) {
    for (int j = 0; j < H_DIM; ++j)
      acc = fmaf(W1[k * H_DIM + j], W2[j * T_STEPS + n], acc);
  }
  WcT[idx] = (_Float16)acc;
  if (k == 0) {
    float s = 0.0f;
    if (n < T_STEPS) {
      for (int j = 0; j < H_DIM; ++j)
        s = fmaf(b1[j], W2[j * T_STEPS + n], s);
      s += b2[n];
    }
    bc[n] = s;
  }
}

// ---------------------------------------------------------------------------
// Kernel 1: fused  logits = X @ Wc + bc  (WMMA f16->f32)  ->  sigmoid  ->
//           IDM(leader) / IDM(mid) mixture -> act_seq, att_scores.
// Block = 256 threads = 8 waves; wave owns 16 rows; 3 N-tiles x 4 K-chunks.
// ---------------------------------------------------------------------------
__global__ __launch_bounds__(256)
void idm_fused_kernel(const float* __restrict__ s_in, const float* __restrict__ prm,
                      const float* __restrict__ dec,
                      const _Float16* __restrict__ WcT_g, const float* __restrict__ bc_g,
                      float* __restrict__ act_out, float* __restrict__ att_out, int B) {
  __shared__ _Float16 WcT[N_PAD * K_PAD];   // 13056 bytes

  // Stage fused weights into LDS (816 x 16B).
  {
    const v4u* __restrict__ src = (const v4u*)WcT_g;
    v4u* dst = (v4u*)WcT;
    for (int i = threadIdx.x; i < (N_PAD * K_PAD) / 8; i += 256)
      dst[i] = src[i];
  }
  __syncthreads();

  const int wave = threadIdx.x >> 5;
  const int lane = threadIdx.x & 31;
  const int l16  = lane & 15;
  const int hi   = lane >> 4;                  // lane group (0: lanes 0-15, 1: 16-31)

  const int rowbase = blockIdx.x * 128 + wave * 16;
  int arow = rowbase + l16;
  if (arow >= B) arow = B - 1;                 // safe clamp (B is a multiple of 128)

  // ---- A fragments: 16x32 f16 per K-chunk, CDNA5 layout:
  //      lane<16: K = +0..7, +16..23 ; lane>=16: K = +8..15, +24..31 ; M = lane%16
  v16h afrag[4];
  const float* __restrict__ ap = dec + (long)arow * D_IN;
#pragma unroll
  for (int kc = 0; kc < 4; ++kc) {
    const int kb = kc * 32 + hi * 8;
    v4f x0 = *(const v4f*)(ap + kb);
    v4f x1 = *(const v4f*)(ap + kb + 4);
    v4f x2 = *(const v4f*)(ap + kb + 16);
    v4f x3 = *(const v4f*)(ap + kb + 20);
    v16h a;
#pragma unroll
    for (int i = 0; i < 4; ++i) {
      a[i]      = (_Float16)x0[i];
      a[4 + i]  = (_Float16)x1[i];
      a[8 + i]  = (_Float16)x2[i];
      a[12 + i] = (_Float16)x3[i];
    }
    afrag[kc] = a;
  }

  // ---- GEMM: 3 N-tiles, K = 4 chunks of 32.  B-frag 32x16 f16 layout:
  //      lane<16: K = +0..7, +16..23 ; lane>=16: K = +8..15, +24..31 ; N = lane%16.
  v8f acc[3];
#pragma unroll
  for (int nt = 0; nt < 3; ++nt) {
    v8f c = {0.f, 0.f, 0.f, 0.f, 0.f, 0.f, 0.f, 0.f};
    const _Float16* bp = &WcT[(nt * 16 + l16) * K_PAD];
#pragma unroll
    for (int kc = 0; kc < 4; ++kc) {
      const int kb = kc * 32 + hi * 8;
      v8h blo = *(const v8h*)(bp + kb);        // 16B-aligned ds_load_b128
      v8h bhi = *(const v8h*)(bp + kb + 16);
      v16h bf = __builtin_shufflevector(blo, bhi,
                 0,1,2,3,4,5,6,7,8,9,10,11,12,13,14,15);
      c = __builtin_amdgcn_wmma_f32_16x16x32_f16(false, afrag[kc], false, bf,
                                                 (short)0, c, false, false);
    }
    acc[nt] = c;
  }

  float bcv[3];
#pragma unroll
  for (int nt = 0; nt < 3; ++nt) bcv[nt] = bc_g[nt * 16 + l16];

  // ---- Fused epilogue.  D layout: VGPR r -> M = r (lanes 0-15) / r+8 (16-31),
  //      N = lane%16.  t (time step) == output column.
#pragma unroll
  for (int r = 0; r < 8; ++r) {
    const int row = rowbase + hi * 8 + r;
    if (row >= B) continue;
    const float* __restrict__ pp = prm + (long)row * 5;
    const float p0 = pp[0], p1 = pp[1], p2 = pp[2], p3 = pp[3], p4 = pp[4];
    const float c2     = 0.5f / sqrtf(p3 * p4);   // 1 / (2*sqrt(max_act*min_act))
    const float inv_p0 = 1.0f / p0;
    const float* __restrict__ srow = s_in + (long)row * (T_STEPS * 7);
    const long obase = (long)row * T_STEPS;
#pragma unroll
    for (int nt = 0; nt < 3; ++nt) {
      const int t = nt * 16 + l16;
      if (t < T_STEPS) {
        const float logit = acc[nt][r] + bcv[nt];
        const float att   = 1.0f / (1.0f + expf(-ATT_TEMP * logit));
        const float* sv = srow + t * 7;
        const float vel = sv[0], dvl = sv[2], dxl = sv[3], dvm = sv[5], dxm = sv[6];
        const float vr  = vel * inv_p0;
        const float vr2 = vr * vr;
        const float v4t = vr2 * vr2;               // (vel/desired_v)^4
        const float gb  = p2 + p1 * vel;
        const float gl  = gb + vel * dvl * c2;     // desired_gap (leader)
        const float gm  = gb + vel * dvm * c2;     // desired_gap (mid)
        const float ql  = gl / dxl, qm = gm / dxm;
        float fl = p3 * (1.0f - v4t - ql * ql);
        float fm = p3 * (1.0f - v4t - qm * qm);
        fl = fminf(fmaxf(fl, -ACT_CLIP), ACT_CLIP);
        fm = fminf(fmaxf(fm, -ACT_CLIP), ACT_CLIP);
        act_out[obase + t] = att * fl + (1.0f - att) * fm;
        att_out[obase + t] = att;
      }
    }
  }
}

// ---------------------------------------------------------------------------
extern "C" void kernel_launch(void* const* d_in, const int* in_sizes, int n_in,
                              void* d_out, int out_size, void* d_ws, size_t ws_size,
                              hipStream_t stream) {
  const float* s_in = (const float*)d_in[0];   // (B, 40, 7)
  const float* prm  = (const float*)d_in[1];   // (B, 5)
  const float* dec  = (const float*)d_in[2];   // (B, 128)
  const float* W1   = (const float*)d_in[3];   // (128, 100)
  const float* b1   = (const float*)d_in[4];   // (100,)
  const float* W2   = (const float*)d_in[5];   // (100, 40)
  const float* b2   = (const float*)d_in[6];   // (40,)

  const int B = in_sizes[1] / 5;

  _Float16* WcT = (_Float16*)d_ws;                                   // 13056 B
  float*    bc  = (float*)((char*)d_ws + N_PAD * K_PAD * sizeof(_Float16)); // +192 B

  float* act_out = (float*)d_out;                 // first B*T floats
  float* att_out = act_out + (size_t)B * T_STEPS; // second B*T floats

  const int prep_blocks = (N_PAD * K_PAD + 255) / 256;   // 26
  idm_prep_weights<<<prep_blocks, 256, 0, stream>>>(W1, b1, W2, b2, WcT, bc);

  const int blocks = (B + 127) / 128;                     // 2048
  idm_fused_kernel<<<blocks, 256, 0, stream>>>(s_in, prm, dec, WcT, bc,
                                               act_out, att_out, B);
}
